// GlmOcrVisionAttention_33835752358493
// MI455X (gfx1250) — matrix operand order repaired and checked by reading, not compiled, not run on hardware
//
#include <hip/hip_runtime.h>
#include <hip/hip_bf16.h>

#define SEQ   2048
#define EMB   1536
#define NH    12
#define HD    128
#define F3    (3 * EMB)

typedef __attribute__((ext_vector_type(16))) _Float16 v16h;
typedef __attribute__((ext_vector_type(8)))  float    v8f;

__device__ __forceinline__ int lane_id() { return (int)(threadIdx.x & 31); }

// ---------------------------------------------------------------------------
// WMMA fragment loaders (this code's contiguous k order, same for A and B, wave32)
// A-matrix 16x32 f16: lanes 0-15 row M=lane hold K 0..15;
//                     lanes 16-31 row M=lane-16 hold K 16..31
__device__ __forceinline__ v16h load_a_frag(const _Float16* base, int ld) {
  int lane = lane_id();
  const _Float16* p = base + (size_t)(lane & 15) * ld + ((lane & 16) ? 16 : 0);
  v16h r;
#pragma unroll
  for (int i = 0; i < 8; ++i) r[i] = p[i];
#pragma unroll
  for (int i = 0; i < 8; ++i) r[i + 8] = p[i + 8];
  return r;
}

// B-matrix 32x16 f16 loaded from B^T stored row-major [N][K]:
// lanes 0-15 col N=lane hold K 0..15; lanes 16-31 col N=lane-16 hold K 16..31
__device__ __forceinline__ v16h load_b_frag(const _Float16* baseT, int ld) {
  int lane = lane_id();
  const _Float16* p = baseT + (size_t)(lane & 15) * ld + ((lane & 16) ? 16 : 0);
  v16h r;
#pragma unroll
  for (int i = 0; i < 16; ++i) r[i] = p[i];
  return r;
}

__device__ __forceinline__ v8f wmma16(v16h a, v16h b, v8f c) {
  return __builtin_amdgcn_wmma_f32_16x16x32_f16(false, a, false, b, (short)0, c,
                                                false, false);
}

// gfx1250 async global->LDS DMA (ASYNCcnt-tracked), 16B per lane.
__device__ __forceinline__ void async_load_b128(uint32_t lds_addr, const void* gaddr) {
  asm volatile("global_load_async_to_lds_b128 %0, %1, off"
               :
               : "v"(lds_addr), "v"((uint64_t)(uintptr_t)gaddr)
               : "memory");
}
__device__ __forceinline__ void wait_async_le0() {
  asm volatile("s_wait_asynccnt 0x0" ::: "memory");
}
__device__ __forceinline__ void wait_async_le2() {
  asm volatile("s_wait_asynccnt 0x2" ::: "memory");
}

// ---------------------------------------------------------------------------
// Elementwise conversion f32 -> f16
__global__ __launch_bounds__(256) void k_convert_f16(const float* __restrict__ x,
                                                     _Float16* __restrict__ y,
                                                     size_t n) {
  size_t i = (size_t)blockIdx.x * blockDim.x + threadIdx.x;
  if (i < n) y[i] = (_Float16)x[i];
}

// Transpose W[E][F] (f32) -> WT[F][E] (f16)
__global__ __launch_bounds__(256) void k_transpose_f16(const float* __restrict__ w,
                                                       _Float16* __restrict__ wt,
                                                       int E, int F) {
  size_t i = (size_t)blockIdx.x * blockDim.x + threadIdx.x;
  if (i >= (size_t)E * F) return;
  int f = (int)(i % F);
  int e = (int)(i / F);
  wt[(size_t)f * E + e] = (_Float16)w[i];
}

// ---------------------------------------------------------------------------
// C[M][N] f32 = A[M][K] f16 x BT[N][K] f16 + bias[N]
// grid = (N/256, M/64), block = 128 (4 waves).
// Block computes a 64x256 C tile. The 64x32 A k-slab is DMA'd into LDS with
// GLOBAL_LOAD_ASYNC_TO_LDS_B128, double-buffered so the DMA for slab k+32
// hides behind the 16 WMMAs of slab k. Each wave owns a 64x64 register tile.
#define APAD 40                       // LDS row stride in halves (80B, conflict-free)
#define ABUF (64 * APAD * 2)          // bytes per A buffer
__global__ __launch_bounds__(128) void k_gemm_wmma(const _Float16* __restrict__ A,
                                                   const _Float16* __restrict__ BT,
                                                   const float* __restrict__ bias,
                                                   float* __restrict__ C,
                                                   int M, int N, int K) {
  __shared__ _Float16 tA[2][64 * APAD];  // 2 x 5 KB

  int wave = threadIdx.x >> 5;
  int lane = lane_id();
  int tid = (int)threadIdx.x;  // 0..127
  int m0 = blockIdx.y * 64;
  int n0 = blockIdx.x * 256 + wave * 64;

  // DMA addressing: lane moves 16B; row = tid/4 (0..31), col-halves = (tid%4)*8.
  int ar = tid >> 2;
  int ac = (tid & 3) * 8;
  uint32_t ldsbase = (uint32_t)(uintptr_t)(void*)&tA[0][0];
  uint32_t l0 = ldsbase + (uint32_t)(ar * APAD + ac) * 2u;
  uint32_t l1 = ldsbase + (uint32_t)((ar + 32) * APAD + ac) * 2u;
  const _Float16* g0 = A + (size_t)(m0 + ar) * K + ac;
  const _Float16* g1 = A + (size_t)(m0 + 32 + ar) * K + ac;

  v8f acc[4][4];
#pragma unroll
  for (int mi = 0; mi < 4; ++mi)
#pragma unroll
    for (int ni = 0; ni < 4; ++ni) acc[mi][ni] = (v8f){};

  // prologue: stage slab k=0 into buffer 0
  async_load_b128(l0, g0);
  async_load_b128(l1, g1);

  int cur = 0;
  for (int k = 0; k < K; k += 32) {
    if (k + 32 < K) {  // prefetch next slab into the other buffer
      uint32_t nb = (uint32_t)(cur ^ 1) * ABUF;
      async_load_b128(l0 + nb, g0 + k + 32);
      async_load_b128(l1 + nb, g1 + k + 32);
      wait_async_le2();  // in-order completion => current slab has landed
    } else {
      wait_async_le0();
    }
    __syncthreads();

    const _Float16* tcur = &tA[cur][0];
    v16h af[4], bf[4];
#pragma unroll
    for (int mi = 0; mi < 4; ++mi) af[mi] = load_a_frag(tcur + mi * 16 * APAD, APAD);
#pragma unroll
    for (int ni = 0; ni < 4; ++ni)
      bf[ni] = load_b_frag(BT + (size_t)(n0 + ni * 16) * K + k, K);
#pragma unroll
    for (int mi = 0; mi < 4; ++mi)
#pragma unroll
      for (int ni = 0; ni < 4; ++ni)
        acc[mi][ni] = wmma16(af[mi], bf[ni], acc[mi][ni]);

    __syncthreads();  // all waves done reading before this buffer is re-DMA'd
    cur ^= 1;
  }

  int nc = lane & 15;
  int mb = (lane & 16) ? 8 : 0;
#pragma unroll
  for (int ni = 0; ni < 4; ++ni) {
    float bv = bias[n0 + ni * 16 + nc];
#pragma unroll
    for (int mi = 0; mi < 4; ++mi)
#pragma unroll
      for (int r = 0; r < 8; ++r)
        C[(size_t)(m0 + mi * 16 + mb + r) * N + (n0 + ni * 16 + nc)] =
            acc[mi][ni][r] + bv;
  }
}

// ---------------------------------------------------------------------------
// Per (token, head): RMS-norm q,k (d=128) + rotary (d2=64), repack head-major.
// qh,kh: [NH][SEQ][HD] f16 ; vT: [NH][HD][SEQ] f16
__global__ __launch_bounds__(128) void k_qkv_post(const float* __restrict__ qkv,
                                                  const float* __restrict__ cosb,
                                                  const float* __restrict__ sinb,
                                                  const float* __restrict__ qw,
                                                  const float* __restrict__ kw,
                                                  _Float16* __restrict__ qh,
                                                  _Float16* __restrict__ kh,
                                                  _Float16* __restrict__ vT) {
  int tok = blockIdx.x;
  int head = blockIdx.y;
  int d = threadIdx.x;  // 0..127
  const float* base = qkv + (size_t)tok * F3 + (size_t)head * HD;
  float qv = base[d];
  float kv = base[EMB + d];
  float vv = base[2 * EMB + d];

  __shared__ float red[2][4];
  __shared__ float qn[HD];
  __shared__ float kn[HD];

  float sq = qv * qv, sk = kv * kv;
#pragma unroll
  for (int off = 16; off > 0; off >>= 1) {
    sq += __shfl_xor(sq, off, 32);
    sk += __shfl_xor(sk, off, 32);
  }
  int wave = threadIdx.x >> 5;
  if ((threadIdx.x & 31) == 0) { red[0][wave] = sq; red[1][wave] = sk; }
  __syncthreads();
  float sumq = red[0][0] + red[0][1] + red[0][2] + red[0][3];
  float sumk = red[1][0] + red[1][1] + red[1][2] + red[1][3];
  float rq = rsqrtf(sumq * (1.0f / HD) + 1e-5f);
  float rk = rsqrtf(sumk * (1.0f / HD) + 1e-5f);
  qn[d] = qv * rq * qw[d];
  kn[d] = kv * rk * kw[d];
  __syncthreads();

  float c = cosb[(size_t)tok * 64 + (d & 63)];
  float s = sinb[(size_t)tok * 64 + (d & 63)];
  float q1, k1;
  if (d < 64) {
    q1 = qn[d] * c - qn[d + 64] * s;
    k1 = kn[d] * c - kn[d + 64] * s;
  } else {
    q1 = qn[d] * c + qn[d - 64] * s;
    k1 = kn[d] * c + kn[d - 64] * s;
  }
  size_t o = ((size_t)head * SEQ + tok) * HD + d;
  qh[o] = (_Float16)q1;
  kh[o] = (_Float16)k1;
  vT[((size_t)head * HD + d) * SEQ + tok] = (_Float16)vv;
}

// ---------------------------------------------------------------------------
// Flash attention. grid = (SEQ/64, NH), block = 128 (4 waves); each wave owns a
// 16-row q tile. The 32-key K/V slabs are async-DMA'd into LDS once per block
// and shared by all 4 waves (4x less L1 traffic than per-wave loads).
#define KPAD 136  // K tile row stride (halves): 68 words, conflict-free, 16B-aligned
#define VPAD 40   // V tile row stride (halves): 20 words, conflict-free, 16B-aligned
__global__ __launch_bounds__(128) void k_attn_wmma(const _Float16* __restrict__ qh,
                                                   const _Float16* __restrict__ kh,
                                                   const _Float16* __restrict__ vT,
                                                   const int* __restrict__ cu,
                                                   int nseg,
                                                   _Float16* __restrict__ ctx) {
  __shared__ _Float16 kt[32 * KPAD];   // 8.5 KB: 32 keys x 128 d
  __shared__ _Float16 vt[128 * VPAD];  // 10 KB: 128 d x 32 keys
  __shared__ _Float16 pls[4][16][32];  // wave-private P re-layout staging

  int head = blockIdx.y;
  int wave = threadIdx.x >> 5;
  int lane = lane_id();
  int tid = (int)threadIdx.x;
  int q0 = blockIdx.x * 64 + wave * 16;

  int lo = 0, hi = SEQ;
  for (int s = 0; s < nseg; ++s) {
    int a = cu[s], b = cu[s + 1];
    if (q0 >= a && q0 < b) { lo = a; hi = b; }
  }

  const _Float16* qbase = qh + ((size_t)head * SEQ + q0) * HD;
  const _Float16* kbase = kh + (size_t)head * SEQ * HD;
  const _Float16* vbase = vT + (size_t)head * HD * SEQ;

  // DMA addressing (16B per lane):
  // K tile: 32 rows x 256B -> 16 lanes/row, 8 rows/issue, 4 issues
  // V tile: 128 rows x 64B -> 4 lanes/row, 32 rows/issue, 4 issues
  uint32_t ktb = (uint32_t)(uintptr_t)(void*)kt;
  uint32_t vtb = (uint32_t)(uintptr_t)(void*)vt;
  uint32_t klds = ktb + (uint32_t)((tid >> 4) * KPAD + (tid & 15) * 8) * 2u;
  uint32_t vlds = vtb + (uint32_t)((tid >> 2) * VPAD + (tid & 3) * 8) * 2u;
  const _Float16* kg = kbase + (size_t)(tid >> 4) * HD + (tid & 15) * 8;
  const _Float16* vg = vbase + (size_t)(tid >> 2) * SEQ + (tid & 3) * 8;

  v16h qf[4];
#pragma unroll
  for (int c = 0; c < 4; ++c) qf[c] = load_a_frag(qbase + c * 32, HD);

  v8f o[8];
#pragma unroll
  for (int d = 0; d < 8; ++d) o[d] = (v8f){};
  float mrow[8], lrow[8];
#pragma unroll
  for (int r = 0; r < 8; ++r) { mrow[r] = -3.0e38f; lrow[r] = 0.0f; }

  _Float16* pb = &pls[wave][0][0];
  int nidx = lane & 15;
  int mb = (lane & 16) ? 8 : 0;
  const float scale = 0.08838834764831845f;  // 128^-0.5

  for (int k0 = lo; k0 < hi; k0 += 32) {
#pragma unroll
    for (int j = 0; j < 4; ++j) {
      async_load_b128(klds + (uint32_t)(j * 8 * KPAD * 2),
                      kg + (size_t)(k0 + j * 8) * HD);
      async_load_b128(vlds + (uint32_t)(j * 32 * VPAD * 2),
                      vg + (size_t)(j * 32) * SEQ + k0);
    }
    wait_async_le0();
    __syncthreads();

    // S tiles: 16 q-rows x 32 keys (two 16x16 WMMA accumulators)
    v8f s0 = {}, s1 = {};
#pragma unroll
    for (int c = 0; c < 4; ++c) {
      v16h kf0 = load_b_frag(&kt[c * 32], KPAD);
      v16h kf1 = load_b_frag(&kt[16 * KPAD + c * 32], KPAD);
      s0 = wmma16(qf[c], kf0, s0);
      s1 = wmma16(qf[c], kf1, s1);
    }
    float sv0[8], sv1[8];
#pragma unroll
    for (int r = 0; r < 8; ++r) {
      float a0 = s0[r] * scale;
      float a1 = s1[r] * scale;
      if (k0 + nidx >= hi)      a0 = -3.0e38f;
      if (k0 + 16 + nidx >= hi) a1 = -3.0e38f;
      sv0[r] = a0; sv1[r] = a1;
    }
    float mnew[8];
#pragma unroll
    for (int r = 0; r < 8; ++r) mnew[r] = fmaxf(sv0[r], sv1[r]);
#pragma unroll
    for (int off = 1; off <= 8; off <<= 1) {
#pragma unroll
      for (int r = 0; r < 8; ++r)
        mnew[r] = fmaxf(mnew[r], __shfl_xor(mnew[r], off, 32));
    }
    float alpha[8];
#pragma unroll
    for (int r = 0; r < 8; ++r) {
      float mn = fmaxf(mrow[r], mnew[r]);
      alpha[r] = __expf(mrow[r] - mn);
      mrow[r] = mn;
      lrow[r] *= alpha[r];
    }
#pragma unroll
    for (int d = 0; d < 8; ++d)
#pragma unroll
      for (int r = 0; r < 8; ++r) o[d][r] *= alpha[r];

    float ls[8];
#pragma unroll
    for (int r = 0; r < 8; ++r) {
      float p0 = __expf(sv0[r] - mrow[r]);
      float p1 = __expf(sv1[r] - mrow[r]);
      ls[r] = p0 + p1;
      pb[(mb + r) * 32 + nidx]      = (_Float16)p0;
      pb[(mb + r) * 32 + 16 + nidx] = (_Float16)p1;
    }
#pragma unroll
    for (int off = 1; off <= 8; off <<= 1) {
#pragma unroll
      for (int r = 0; r < 8; ++r) ls[r] += __shfl_xor(ls[r], off, 32);
    }
#pragma unroll
    for (int r = 0; r < 8; ++r) lrow[r] += ls[r];

    // per-wave DS ops are in-order: safe to read back our private slice
    v16h pf = load_a_frag(pb, 32);
#pragma unroll
    for (int d = 0; d < 8; ++d) {
      v16h vf = load_b_frag(&vt[(size_t)(d * 16) * VPAD], VPAD);
      o[d] = wmma16(pf, vf, o[d]);
    }

    __syncthreads();  // all waves done with kt/vt before next slab's DMA
  }

  float inv[8];
#pragma unroll
  for (int r = 0; r < 8; ++r) inv[r] = 1.0f / lrow[r];
#pragma unroll
  for (int d = 0; d < 8; ++d)
#pragma unroll
    for (int r = 0; r < 8; ++r)
      ctx[(size_t)(q0 + mb + r) * EMB + head * HD + d * 16 + nidx] =
          (_Float16)(o[d][r] * inv[r]);
}

// ---------------------------------------------------------------------------
extern "C" void kernel_launch(void* const* d_in, const int* in_sizes, int n_in,
                              void* d_out, int out_size, void* d_ws, size_t ws_size,
                              hipStream_t stream) {
  const float* x      = (const float*)d_in[0];
  const int*   cu     = (const int*)d_in[1];
  const float* cosb   = (const float*)d_in[2];
  const float* sinb   = (const float*)d_in[3];
  const float* w_qkv  = (const float*)d_in[4];
  const float* b_qkv  = (const float*)d_in[5];
  const float* qnw    = (const float*)d_in[6];
  const float* knw    = (const float*)d_in[7];
  const float* w_proj = (const float*)d_in[8];
  const float* b_proj = (const float*)d_in[9];
  float* out = (float*)d_out;
  int nseg = in_sizes[1] - 1;

  char* ws = (char*)d_ws;
  size_t off = 0;
  auto alloc = [&](size_t bytes) -> void* {
    void* p = ws + off;
    off += (bytes + 255) & ~(size_t)255;
    return p;
  };
  _Float16* x_h    = (_Float16*)alloc((size_t)SEQ * EMB * 2);
  _Float16* wqkvT  = (_Float16*)alloc((size_t)F3 * EMB * 2);
  _Float16* wprT   = (_Float16*)alloc((size_t)EMB * EMB * 2);
  float*    qkv    = (float*)   alloc((size_t)SEQ * F3 * 4);
  _Float16* qh     = (_Float16*)alloc((size_t)NH * SEQ * HD * 2);
  _Float16* kh     = (_Float16*)alloc((size_t)NH * SEQ * HD * 2);
  _Float16* vT     = (_Float16*)alloc((size_t)NH * SEQ * HD * 2);
  _Float16* ctxh   = (_Float16*)alloc((size_t)SEQ * EMB * 2);

  {
    size_t n = (size_t)SEQ * EMB;
    k_convert_f16<<<(unsigned)((n + 255) / 256), 256, 0, stream>>>(x, x_h, n);
  }
  {
    size_t n = (size_t)EMB * F3;
    k_transpose_f16<<<(unsigned)((n + 255) / 256), 256, 0, stream>>>(w_qkv, wqkvT, EMB, F3);
  }
  {
    size_t n = (size_t)EMB * EMB;
    k_transpose_f16<<<(unsigned)((n + 255) / 256), 256, 0, stream>>>(w_proj, wprT, EMB, EMB);
  }
  k_gemm_wmma<<<dim3(F3 / 256, SEQ / 64), 128, 0, stream>>>(x_h, wqkvT, b_qkv, qkv,
                                                            SEQ, F3, EMB);
  k_qkv_post<<<dim3(SEQ, NH), 128, 0, stream>>>(qkv, cosb, sinb, qnw, knw, qh, kh, vT);
  k_attn_wmma<<<dim3(SEQ / 64, NH), 128, 0, stream>>>(qh, kh, vT, cu, nseg, ctxh);
  k_gemm_wmma<<<dim3(EMB / 256, SEQ / 64), 128, 0, stream>>>(ctxh, wprT, b_proj, out,
                                                             SEQ, EMB, EMB);
}
